// DecoderLayer_14439680049318
// MI455X (gfx1250) — compile-verified, matrix operand order checked by reading, not run on hardware
//
#include <hip/hip_runtime.h>
#include <math.h>

// ---------------------------------------------------------------------------
// MI455X (gfx1250) decoder layer: LN -> QKV (bf16 WMMA GEMM) -> flash attn
// (bf16 WMMA) -> out-proj + residual -> LN -> FFN(GELU) -> residual.
// All matrix math through v_wmma_f32_16x16x32_bf16 (fp32 accumulate).
// GEMM uses double-buffered LDS staging via async global->LDS loads
// (ASYNCcnt) when the toolchain exposes the gfx1250 builtin.
// ---------------------------------------------------------------------------

typedef __bf16 bf16_t;
typedef __attribute__((ext_vector_type(16))) __bf16 v16bf;
typedef __attribute__((ext_vector_type(8)))  float  v8f;

#define B_   4
#define S_   2048
#define D_   1024
#define H_   16
#define DK_  64
#define MTOT (B_ * S_)      // 8192

#if __has_builtin(__builtin_amdgcn_global_load_async_to_lds_b128)
#define USE_ASYNC_LDS 1
#endif

static __device__ __forceinline__ bf16_t f2bf(float f) {
  unsigned u = __builtin_bit_cast(unsigned, f);
  u = (u + 0x7fffu + ((u >> 16) & 1u)) >> 16;   // round-to-nearest-even
  unsigned short h = (unsigned short)u;
  return __builtin_bit_cast(bf16_t, h);
}

static __device__ __forceinline__ v16bf make_frag(uint4 lo, uint4 hi) {
  union { uint4 u[2]; v16bf v; } x;
  x.u[0] = lo; x.u[1] = hi;
  return x.v;
}

static __device__ __forceinline__ v8f wmma_bf16(v16bf a, v16bf b, v8f c) {
  return __builtin_amdgcn_wmma_f32_16x16x32_bf16(false, a, false, b,
                                                 (short)0, c, false, false);
}

#ifdef USE_ASYNC_LDS
// Builtin signature (from clang diagnostic): param types are pointers to
// GCC-vector 'int __attribute__((vector_size(16)))' in AS1 (global) / AS3 (LDS).
typedef int v4i_t __attribute__((vector_size(16)));
typedef v4i_t __attribute__((address_space(1))) gv4i_t;
typedef v4i_t __attribute__((address_space(3))) lv4i_t;

static __device__ __forceinline__ void wait_async0() {
#if __has_builtin(__builtin_amdgcn_s_wait_asynccnt)
  __builtin_amdgcn_s_wait_asynccnt(0);
#else
  asm volatile("s_wait_asynccnt 0x0" ::: "memory");
#endif
}
// 16B global -> LDS, no VGPR data path, tracked by ASYNCcnt.
#define STAGE_CP(dst, src)                                              \
  __builtin_amdgcn_global_load_async_to_lds_b128(                       \
      (gv4i_t*)(v4i_t*)(src), (lv4i_t*)(v4i_t*)(dst), 0, 0)
#else
#define STAGE_CP(dst, src) (*(uint4*)(dst) = *(const uint4*)(src))
#endif

// ---------------------------------------------------------------------------
// Weight transpose + fp32 -> bf16 convert:  Wt[n][k] = (bf16) W[k][n]
// ---------------------------------------------------------------------------
__global__ __launch_bounds__(256) void transpose_w_bf16(
    const float* __restrict__ W, bf16_t* __restrict__ Wt) {
  __shared__ float tile[32][33];
  const int K = 1024, N = 1024;
  int n0 = blockIdx.x * 32, k0 = blockIdx.y * 32;
  int tx = threadIdx.x, ty = threadIdx.y;       // (32, 8)
  #pragma unroll
  for (int i = ty; i < 32; i += 8)
    tile[i][tx] = W[(size_t)(k0 + i) * N + n0 + tx];
  __syncthreads();
  #pragma unroll
  for (int i = ty; i < 32; i += 8)
    Wt[(size_t)(n0 + i) * K + k0 + tx] = f2bf(tile[tx][i]);
}

// ---------------------------------------------------------------------------
// V transpose per head:  vt[bh][dk][s] = v[bh][s][dk]   (bf16)
// ---------------------------------------------------------------------------
__global__ __launch_bounds__(256) void transpose_v_bf16(
    const bf16_t* __restrict__ v, bf16_t* __restrict__ vt) {
  __shared__ bf16_t tile[32][33];
  int bh = blockIdx.z;
  int s0 = blockIdx.x * 32, d0 = blockIdx.y * 32;
  int tx = threadIdx.x, ty = threadIdx.y;       // (32, 8)
  size_t base = (size_t)bh * (S_ * DK_);
  #pragma unroll
  for (int i = ty; i < 32; i += 8)
    tile[i][tx] = v[base + (size_t)(s0 + i) * DK_ + d0 + tx];
  __syncthreads();
  #pragma unroll
  for (int i = ty; i < 32; i += 8)
    vt[base + (size_t)(d0 + i) * S_ + s0 + tx] = tile[tx][i];
}

// ---------------------------------------------------------------------------
// LayerNorm (torch-style: unbiased std, (x-mean)/(std+eps)*alpha+beta) -> bf16
// One 256-thread block per row of 1024.
// ---------------------------------------------------------------------------
__global__ __launch_bounds__(256) void layernorm_bf16(
    const float* __restrict__ x, const float* __restrict__ alpha,
    const float* __restrict__ beta, bf16_t* __restrict__ out) {
  __shared__ float red1[8], red2[8];
  int row = blockIdx.x;
  int tid = threadIdx.x, wave = tid >> 5, lane = tid & 31;
  const float* xr = x + (size_t)row * D_;
  float v[4]; float s = 0.f;
  #pragma unroll
  for (int i = 0; i < 4; ++i) { v[i] = xr[tid + i * 256]; s += v[i]; }
  #pragma unroll
  for (int off = 1; off < 32; off <<= 1) s += __shfl_xor(s, off, 32);
  if (lane == 0) red1[wave] = s;
  __syncthreads();
  float tot = 0.f;
  #pragma unroll
  for (int w = 0; w < 8; ++w) tot += red1[w];
  float mean = tot * (1.0f / 1024.0f);
  float vs = 0.f;
  #pragma unroll
  for (int i = 0; i < 4; ++i) { float d = v[i] - mean; vs += d * d; }
  #pragma unroll
  for (int off = 1; off < 32; off <<= 1) vs += __shfl_xor(vs, off, 32);
  if (lane == 0) red2[wave] = vs;
  __syncthreads();
  float tv = 0.f;
  #pragma unroll
  for (int w = 0; w < 8; ++w) tv += red2[w];
  float std_ = sqrtf(tv * (1.0f / 1023.0f));
  float inv = 1.0f / (std_ + 1e-6f);
  #pragma unroll
  for (int i = 0; i < 4; ++i) {
    int col = tid + i * 256;
    out[(size_t)row * D_ + col] = f2bf(alpha[col] * (v[i] - mean) * inv + beta[col]);
  }
}

// ---------------------------------------------------------------------------
// bf16 WMMA GEMM: C[M,N] = A[M,K] * Bt[N,K]^T (+bias, + epilogue)
// M=8192, N=K=1024. Block tile 128x128, BK=32, 8 waves each own 32x64.
// Double-buffered LDS staging (async global->LDS when available).
// ---------------------------------------------------------------------------
#define BM   128
#define BN   128
#define BK   32
#define LDSK 48      // bf16 elems per LDS row (96 B: 16B-aligned, bank-skewed)

enum { MODE_QKV = 0, MODE_RESID_F32 = 1, MODE_GELU_BF16 = 2 };

__global__ __launch_bounds__(256) void gemm_bf16_wmma(
    const bf16_t* __restrict__ A, const bf16_t* __restrict__ Bt,
    const float* __restrict__ bias, const float* __restrict__ resid,
    float* __restrict__ outF, bf16_t* __restrict__ outB, int mode) {
  __shared__ __align__(16) bf16_t As[2][BM][LDSK];
  __shared__ __align__(16) bf16_t Bs[2][BN][LDSK];
  const int K = 1024, N = 1024;
  int tid = threadIdx.x;
  int wave = tid >> 5, lane = tid & 31;
  int wm = wave >> 1, wn = wave & 1;            // 4 waves in M, 2 in N
  int half = lane >> 4, l16 = lane & 15;
  int kb = half * 8;                            // per-ISA 16-bit A/B K packing
  size_t m0 = (size_t)blockIdx.y * BM;
  size_t n0 = (size_t)blockIdx.x * BN;

  // Branch-free cooperative staging: each thread owns 2 A + 2 B 16B chunks.
  int lrow = tid >> 2;                          // 0..63
  int lcc  = (tid & 3) * 8;                     // chunk col (bf16 elems)
  const bf16_t* gA0 = A  + (m0 + lrow)      * (size_t)K + lcc;
  const bf16_t* gA1 = A  + (m0 + lrow + 64) * (size_t)K + lcc;
  const bf16_t* gB0 = Bt + (n0 + lrow)      * (size_t)K + lcc;
  const bf16_t* gB1 = Bt + (n0 + lrow + 64) * (size_t)K + lcc;

  auto stage = [&](int buf, int k0) {
    STAGE_CP(&As[buf][lrow][lcc],      gA0 + k0);
    STAGE_CP(&As[buf][lrow + 64][lcc], gA1 + k0);
    STAGE_CP(&Bs[buf][lrow][lcc],      gB0 + k0);
    STAGE_CP(&Bs[buf][lrow + 64][lcc], gB1 + k0);
  };

  v8f acc[2][4] = {};
  stage(0, 0);

  for (int k0 = 0, buf = 0; k0 < K; k0 += BK, buf ^= 1) {
#ifdef USE_ASYNC_LDS
    wait_async0();                              // own stage-n copies done
#endif
    __syncthreads();                            // everyone's data ready AND
                                                // stage n-1 compute finished
    if (k0 + BK < K) stage(buf ^ 1, k0 + BK);   // prefetch next stage

    v16bf af[2], bfr[4];
    #pragma unroll
    for (int i = 0; i < 2; ++i) {
      int r = wm * 32 + i * 16 + l16;
      af[i] = make_frag(*(const uint4*)&As[buf][r][kb],
                        *(const uint4*)&As[buf][r][kb + 16]);
    }
    #pragma unroll
    for (int j = 0; j < 4; ++j) {
      int r = wn * 64 + j * 16 + l16;
      bfr[j] = make_frag(*(const uint4*)&Bs[buf][r][kb],
                         *(const uint4*)&Bs[buf][r][kb + 16]);
    }
    #pragma unroll
    for (int i = 0; i < 2; ++i)
      #pragma unroll
      for (int j = 0; j < 4; ++j)
        acc[i][j] = wmma_bf16(af[i], bfr[j], acc[i][j]);
  }

  // Epilogue: C layout -> lane holds row (r + 8*half), col l16 of each tile.
  #pragma unroll
  for (int i = 0; i < 2; ++i) {
    #pragma unroll
    for (int j = 0; j < 4; ++j) {
      #pragma unroll
      for (int r = 0; r < 8; ++r) {
        size_t m = m0 + wm * 32 + i * 16 + r + half * 8;
        size_t n = n0 + wn * 64 + j * 16 + l16;
        float v = acc[i][j][r] + bias[n];
        if (mode == MODE_QKV) {
          size_t b = m >> 11, s = m & 2047;     // S_=2048
          size_t h = n >> 6, dk = n & 63;       // DK_=64
          outB[(((b * H_ + h) * S_ + s) * DK_) + dk] = f2bf(v);
        } else if (mode == MODE_GELU_BF16) {
          float g = 0.5f * v *
              (1.0f + tanhf(0.7978845608028654f * (v + 0.044715f * v * v * v)));
          outB[m * N + n] = f2bf(g);
        } else {                                 // MODE_RESID_F32
          outF[m * N + n] = resid[m * N + n] + v;
        }
      }
    }
  }
}

// ---------------------------------------------------------------------------
// Flash attention, causal. Grid (S/128, B*H). 8 waves, each owns 16 q-rows,
// fully independent (wave-local LDS staging, no block barriers).
// q,k: [bh][s][64] bf16; vt: [bh][64][s] bf16; out: [b][s][1024] bf16.
// ---------------------------------------------------------------------------
__global__ __launch_bounds__(256) void attn_fwd(
    const bf16_t* __restrict__ q, const bf16_t* __restrict__ k,
    const bf16_t* __restrict__ vt, bf16_t* __restrict__ attn_out) {
  __shared__ __align__(16) bf16_t psh[8][16][32];   // per-wave P staging
  int bh = blockIdx.y;
  int qbase = blockIdx.x * 128;
  int tid = threadIdx.x, wave = tid >> 5, lane = tid & 31;
  int half = lane >> 4, l16 = lane & 15;
  int kb = half * 8;
  size_t base = (size_t)bh * (S_ * DK_);

  int qrow = qbase + wave * 16 + l16;
  const bf16_t* qp = q + base + (size_t)qrow * DK_;
  v16bf qf[2];
  qf[0] = make_frag(*(const uint4*)(qp + kb),      *(const uint4*)(qp + kb + 16));
  qf[1] = make_frag(*(const uint4*)(qp + 32 + kb), *(const uint4*)(qp + 32 + kb + 16));

  v8f o[4] = {};
  float mrow[8], lrow[8];
  #pragma unroll
  for (int r = 0; r < 8; ++r) { mrow[r] = -3.0e38f; lrow[r] = 0.f; }

  const int ktiles = (qbase >> 5) + 4;              // causal: keys <= qbase+127
  for (int kt = 0; kt < ktiles; ++kt) {
    // scores S = Q(16x64) @ K_tile^T(64x32), two 16-key sub-tiles
    v8f sc[2] = {};
    #pragma unroll
    for (int t = 0; t < 2; ++t) {
      int krow = kt * 32 + t * 16 + l16;
      const bf16_t* kp = k + base + (size_t)krow * DK_;
      v16bf kf0 = make_frag(*(const uint4*)(kp + kb),      *(const uint4*)(kp + kb + 16));
      v16bf kf1 = make_frag(*(const uint4*)(kp + 32 + kb), *(const uint4*)(kp + 32 + kb + 16));
      sc[t] = wmma_bf16(qf[0], kf0, sc[t]);
      sc[t] = wmma_bf16(qf[1], kf1, sc[t]);
    }
    // scale + causal mask + row max (xor-shuffle over the 16-lane half)
    float tmax[8];
    #pragma unroll
    for (int r = 0; r < 8; ++r) {
      int qg = qbase + wave * 16 + r + half * 8;
      float s0 = sc[0][r] * 0.125f;                 // 1/sqrt(64)
      float s1 = sc[1][r] * 0.125f;
      if (kt * 32 + l16 > qg)      s0 = -1.0e30f;
      if (kt * 32 + 16 + l16 > qg) s1 = -1.0e30f;
      sc[0][r] = s0; sc[1][r] = s1;
      tmax[r] = fmaxf(s0, s1);
    }
    #pragma unroll
    for (int off = 1; off < 16; off <<= 1)
      #pragma unroll
      for (int r = 0; r < 8; ++r)
        tmax[r] = fmaxf(tmax[r], __shfl_xor(tmax[r], off, 32));
    // online softmax update
    float psum[8];
    #pragma unroll
    for (int r = 0; r < 8; ++r) {
      float mnew = fmaxf(mrow[r], tmax[r]);
      float f = __expf(mrow[r] - mnew);
      mrow[r] = mnew;
      float p0 = __expf(sc[0][r] - mnew);
      float p1 = __expf(sc[1][r] - mnew);
      sc[0][r] = p0; sc[1][r] = p1;
      psum[r] = p0 + p1;
      lrow[r] *= f;
      #pragma unroll
      for (int j = 0; j < 4; ++j) o[j][r] *= f;
    }
    #pragma unroll
    for (int off = 1; off < 16; off <<= 1)
      #pragma unroll
      for (int r = 0; r < 8; ++r)
        psum[r] += __shfl_xor(psum[r], off, 32);
    #pragma unroll
    for (int r = 0; r < 8; ++r) lrow[r] += psum[r];

    // P (C layout) -> per-wave LDS -> A fragment. DS ops of one wave are
    // in-order; s_wait_dscnt 0 guarantees the writes landed before reads.
    #pragma unroll
    for (int r = 0; r < 8; ++r) {
      int pr = r + half * 8;
      psh[wave][pr][l16]      = f2bf(sc[0][r]);
      psh[wave][pr][16 + l16] = f2bf(sc[1][r]);
    }
    asm volatile("s_wait_dscnt 0x0" ::: "memory");
    __builtin_amdgcn_wave_barrier();
    v16bf pf = make_frag(*(const uint4*)&psh[wave][l16][kb],
                         *(const uint4*)&psh[wave][l16][kb + 16]);
    // O += P(16x32) @ V(32x64); B operand rows from vt (contiguous)
    #pragma unroll
    for (int j = 0; j < 4; ++j) {
      const bf16_t* vp = vt + base + (size_t)(j * 16 + l16) * S_ + kt * 32;
      v16bf vf = make_frag(*(const uint4*)(vp + kb), *(const uint4*)(vp + kb + 16));
      o[j] = wmma_bf16(pf, vf, o[j]);
    }
  }

  int b = bh >> 4, h = bh & 15;
  #pragma unroll
  for (int r = 0; r < 8; ++r) {
    float inv = 1.0f / lrow[r];
    int srow = qbase + wave * 16 + r + half * 8;
    bf16_t* op = attn_out + ((size_t)(b * S_ + srow)) * D_ + h * DK_;
    #pragma unroll
    for (int j = 0; j < 4; ++j)
      op[j * 16 + l16] = f2bf(o[j][r] * inv);
  }
}

// ---------------------------------------------------------------------------
// Host-side orchestration. Workspace layout (bytes, all 16B-aligned):
//   [0)            6 x 2 MiB : Wt bf16 (Wq,Wk,Wv,Wo,W1,W2 transposed)
//   [12.0 MiB)     16 MiB    : XN bf16 (normed x)    -- reused as ATT
//   [28.0 MiB)     16 MiB    : Q bf16 [b][h][s][dk]  -- reused as XN2
//   [44.0 MiB)     16 MiB    : K bf16 [b][h][s][dk]  -- reused as FFN hidden
//   [60.0 MiB)     16 MiB    : V bf16 [b][h][s][dk]
//   [76.0 MiB)     16 MiB    : VT bf16 [b][h][dk][s]
// Total ~92 MiB.
// ---------------------------------------------------------------------------
extern "C" void kernel_launch(void* const* d_in, const int* in_sizes, int n_in,
                              void* d_out, int out_size, void* d_ws, size_t ws_size,
                              hipStream_t stream) {
  (void)in_sizes; (void)n_in; (void)out_size; (void)ws_size;
  const float* x      = (const float*)d_in[0];
  // d_in[1] = mask (causal tril; implemented directly in attn_fwd)
  const float* Wq = (const float*)d_in[2];  const float* bq = (const float*)d_in[3];
  const float* Wk = (const float*)d_in[4];  const float* bk = (const float*)d_in[5];
  const float* Wv = (const float*)d_in[6];  const float* bv = (const float*)d_in[7];
  const float* Wo = (const float*)d_in[8];  const float* bo = (const float*)d_in[9];
  const float* W1 = (const float*)d_in[10]; const float* b1 = (const float*)d_in[11];
  const float* W2 = (const float*)d_in[12]; const float* b2 = (const float*)d_in[13];
  const float* alpha1 = (const float*)d_in[14]; const float* bias1 = (const float*)d_in[15];
  const float* alpha2 = (const float*)d_in[16]; const float* bias2 = (const float*)d_in[17];
  float* out = (float*)d_out;

  char* ws = (char*)d_ws;
  const size_t WSZ = (size_t)1024 * 1024 * 2;       // one bf16 weight matrix
  const size_t ASZ = (size_t)MTOT * D_ * 2;         // one bf16 activation
  bf16_t* WT0 = (bf16_t*)(ws + 0 * WSZ);
  bf16_t* WT1 = (bf16_t*)(ws + 1 * WSZ);
  bf16_t* WT2 = (bf16_t*)(ws + 2 * WSZ);
  bf16_t* WT3 = (bf16_t*)(ws + 3 * WSZ);
  bf16_t* WT4 = (bf16_t*)(ws + 4 * WSZ);
  bf16_t* WT5 = (bf16_t*)(ws + 5 * WSZ);
  char* act = ws + 6 * WSZ;
  bf16_t* XN  = (bf16_t*)(act + 0 * ASZ);           // normed x; reused as ATT
  bf16_t* Qb  = (bf16_t*)(act + 1 * ASZ);           // reused as XN2
  bf16_t* Kb  = (bf16_t*)(act + 2 * ASZ);           // reused as FFN hidden
  bf16_t* Vb  = (bf16_t*)(act + 3 * ASZ);
  bf16_t* VT  = (bf16_t*)(act + 4 * ASZ);
  bf16_t* ATT = XN;
  bf16_t* XN2 = Qb;
  bf16_t* Hh  = Kb;

  dim3 bW(32, 8), gW(32, 32);
  transpose_w_bf16<<<gW, bW, 0, stream>>>(Wq, WT0);
  transpose_w_bf16<<<gW, bW, 0, stream>>>(Wk, WT1);
  transpose_w_bf16<<<gW, bW, 0, stream>>>(Wv, WT2);
  transpose_w_bf16<<<gW, bW, 0, stream>>>(Wo, WT3);
  transpose_w_bf16<<<gW, bW, 0, stream>>>(W1, WT4);
  transpose_w_bf16<<<gW, bW, 0, stream>>>(W2, WT5);

  layernorm_bf16<<<MTOT, 256, 0, stream>>>(x, alpha1, bias1, XN);

  dim3 gG(1024 / BN, MTOT / BM);                    // (8, 64)
  gemm_bf16_wmma<<<gG, 256, 0, stream>>>(XN, WT0, bq, nullptr, nullptr, Qb, MODE_QKV);
  gemm_bf16_wmma<<<gG, 256, 0, stream>>>(XN, WT1, bk, nullptr, nullptr, Kb, MODE_QKV);
  gemm_bf16_wmma<<<gG, 256, 0, stream>>>(XN, WT2, bv, nullptr, nullptr, Vb, MODE_QKV);

  transpose_v_bf16<<<dim3(S_ / 32, DK_ / 32, B_ * H_), bW, 0, stream>>>(Vb, VT);

  attn_fwd<<<dim3(S_ / 128, B_ * H_), 256, 0, stream>>>(Qb, Kb, VT, ATT);

  // x1 = x + ATT @ Wo + bo   (fp32, written to d_out)
  gemm_bf16_wmma<<<gG, 256, 0, stream>>>(ATT, WT3, bo, x, out, nullptr, MODE_RESID_F32);

  layernorm_bf16<<<MTOT, 256, 0, stream>>>(out, alpha2, bias2, XN2);

  // h = gelu(XN2 @ W1 + b1)
  gemm_bf16_wmma<<<gG, 256, 0, stream>>>(XN2, WT4, b1, nullptr, nullptr, Hh, MODE_GELU_BF16);
  // out = x1 + h @ W2 + b2   (reads+writes d_out in-place per element)
  gemm_bf16_wmma<<<gG, 256, 0, stream>>>(Hh, WT5, b2, out, out, nullptr, MODE_RESID_F32);
}
